// BipartiteGNN_66846870995569
// MI455X (gfx1250) — compile-verified, matrix operand order checked by reading, not run on hardware
//
#include <hip/hip_runtime.h>
#include <stdint.h>

// ---------------------------------------------------------------------------
// CDNA5 (gfx1250) bipartite-GNN pipeline:
//  - v_wmma_f32_16x16x32_bf16 with hi/lo bf16 split (3 WMMAs/K-tile, ~f32 acc)
//  - per-block LDS staging of packed weight fragments via
//    global_load_async_to_lds_b128 (ASYNCcnt), double buffered
//  - v_perm_b32-based bf16 packing for A fragments
// ---------------------------------------------------------------------------

typedef __bf16  bf16_t;
typedef __attribute__((ext_vector_type(16))) __bf16 v16bf;
typedef __attribute__((ext_vector_type(8)))  float  v8f;

#define D128 128
#define FRAGK 4096  // packed bf16 elements per K-tile slab (8 nt * 32 lanes * 16)

__device__ __forceinline__ bf16_t f2bf(float x) {
  unsigned u = __builtin_bit_cast(unsigned, x);
  unsigned r = (u + 0x7FFFu + ((u >> 16) & 1u)) >> 16;
  unsigned short s = (unsigned short)r;
  return __builtin_bit_cast(bf16_t, s);
}
__device__ __forceinline__ float bf2f(bf16_t b) {
  unsigned short s = __builtin_bit_cast(unsigned short, b);
  unsigned u = ((unsigned)s) << 16;
  return __builtin_bit_cast(float, u);
}

__device__ __forceinline__ v8f wmma_bf16(v16bf a, v16bf b, v8f c) {
  // (neg_a, A, neg_b, B, c_mod, C, reuse_a, reuse_b)
  return __builtin_amdgcn_wmma_f32_16x16x32_bf16(false, a, false, b, (short)0, c,
                                                 false, false);
}

// ---- async global -> LDS copy (CDNA5, tracked with ASYNCcnt) ----
__device__ __forceinline__ void async_copy_b128(uint32_t lds_byte_off, uint64_t gaddr) {
  asm volatile("global_load_async_to_lds_b128 %0, %1, off"
               :
               : "v"(lds_byte_off), "v"(gaddr)
               : "memory");
}
__device__ __forceinline__ void wait_async0() {
  asm volatile("s_wait_asynccnt 0x0" ::: "memory");
}

// Stage one K-tile slab of hi+lo packed weights (8KB each) into LDS.
__device__ __forceinline__ void stage_slab(const bf16_t* __restrict__ WH,
                                           const bf16_t* __restrict__ WL, int kt,
                                           bf16_t* dH, bf16_t* dL, int tid) {
  uint32_t lH = (uint32_t)(uintptr_t)(void*)dH + (uint32_t)tid * 32u;
  uint32_t lL = (uint32_t)(uintptr_t)(void*)dL + (uint32_t)tid * 32u;
  uint64_t gH = (uint64_t)(uintptr_t)(WH + (size_t)kt * FRAGK) + (uint64_t)tid * 32u;
  uint64_t gL = (uint64_t)(uintptr_t)(WL + (size_t)kt * FRAGK) + (uint64_t)tid * 32u;
  async_copy_b128(lH, gH);
  async_copy_b128(lH + 16, gH + 16);
  async_copy_b128(lL, gL);
  async_copy_b128(lL + 16, gL + 16);
}

// A fragment as both v16bf (for WMMA) and dwords (for v_perm packing).
union AFrag {
  v16bf v;
  uint32_t u[8];
};

// Fill elements [AT, AT+8) of hi/lo A fragments from 8 contiguous f32.
// hi = trunc16(x) (packed 2-per-dword with v_perm), lo = trunc16(x - hi).
template <int AT>
__device__ __forceinline__ void fill8(AFrag& fh, AFrag& fl, const float* __restrict__ src) {
  const float4* s4 = reinterpret_cast<const float4*>(src);
  float4 a = s4[0], b = s4[1];
  float xs[8] = {a.x, a.y, a.z, a.w, b.x, b.y, b.z, b.w};
#pragma unroll
  for (int p = 0; p < 4; ++p) {
    float x0 = xs[2 * p], x1 = xs[2 * p + 1];
    uint32_t u0 = __builtin_bit_cast(uint32_t, x0);
    uint32_t u1 = __builtin_bit_cast(uint32_t, x1);
    fh.u[AT / 2 + p] = __builtin_amdgcn_perm(u1, u0, 0x07060302u);
    float r0 = x0 - __builtin_bit_cast(float, u0 & 0xffff0000u);
    float r1 = x1 - __builtin_bit_cast(float, u1 & 0xffff0000u);
    fl.u[AT / 2 + p] = __builtin_amdgcn_perm(__builtin_bit_cast(uint32_t, r1),
                                             __builtin_bit_cast(uint32_t, r0),
                                             0x07060302u);
  }
}

// Shared GEMM core: acc[8] += concat[rowA,rowB](16 rows x K) @ W(K x 128).
// Weights streamed through LDS with async copies, double buffered.
template <int KT>
__device__ __forceinline__ void wmma_block_gemm(const float* rowA, const float* rowB,
                                                const bf16_t* __restrict__ WH,
                                                const bf16_t* __restrict__ WL,
                                                bf16_t* smH, bf16_t* smL, v8f acc[8],
                                                int tid) {
  int lane = tid & 31;
  int h = lane >> 4;
  stage_slab(WH, WL, 0, smH, smL, tid);
  wait_async0();
  __syncthreads();
  int buf = 0;
#pragma unroll 1
  for (int kt = 0; kt < KT; ++kt) {
    if (kt + 1 < KT)
      stage_slab(WH, WL, kt + 1, smH + (buf ^ 1) * FRAGK, smL + (buf ^ 1) * FRAGK, tid);
    const float* src = (KT == 8 && kt >= 4) ? rowB : rowA;
    int kh = kt * 32 - ((KT == 8 && kt >= 4) ? 128 : 0) + 8 * h;
    AFrag ah, al;
    fill8<0>(ah, al, src + kh);
    fill8<8>(ah, al, src + kh + 16);
    const bf16_t* bhb = smH + buf * FRAGK;
    const bf16_t* blb = smL + buf * FRAGK;
#pragma unroll
    for (int nt = 0; nt < 8; ++nt) {
      v16bf bh = *reinterpret_cast<const v16bf*>(bhb + (nt * 32 + lane) * 16);
      v16bf bl = *reinterpret_cast<const v16bf*>(blb + (nt * 32 + lane) * 16);
      acc[nt] = wmma_bf16(ah.v, bh, acc[nt]);
      acc[nt] = wmma_bf16(ah.v, bl, acc[nt]);
      acc[nt] = wmma_bf16(al.v, bh, acc[nt]);
    }
    if (kt + 1 < KT) {
      wait_async0();
      __syncthreads();
      buf ^= 1;
    }
  }
}

// ---------------------------------------------------------------------------
// Pack a (K x 128) f32 weight matrix into WMMA B-fragment order, split hi/lo.
// Packed linear index == i + 16*(lane + 32*(nt + 8*kt)).
// ---------------------------------------------------------------------------
__global__ void pack_w_kernel(const float* __restrict__ W, bf16_t* __restrict__ H,
                              bf16_t* __restrict__ Lo, int KT) {
  int idx = blockIdx.x * blockDim.x + threadIdx.x;
  int total = KT * 8 * 32 * 16;
  if (idx >= total) return;
  int i = idx & 15, lane = (idx >> 4) & 31, nt = (idx >> 9) & 7, kt = idx >> 12;
  int h = lane >> 4, n = nt * 16 + (lane & 15);
  int k = kt * 32 + ((i < 8) ? (8 * h + i) : (16 + 8 * h + (i - 8)));
  float x = W[(size_t)k * D128 + n];
  bf16_t hb = f2bf(x);
  H[idx] = hb;
  Lo[idx] = f2bf(x - bf2f(hb));
}

__global__ void zero_kernel(float* __restrict__ p, long long n) {
  long long i = (long long)blockIdx.x * blockDim.x + threadIdx.x;
  long long stride = (long long)gridDim.x * blockDim.x;
  for (; i < n; i += stride) p[i] = 0.0f;
}

// ---------------------------------------------------------------------------
// Edge message kernel: m = relu(concat[Xi[idx_i], Xj[idx_j]] @ W + b),
// scatter-add into aggr[agg_idx]. One wave = 16 edges x 128 output cols.
// ---------------------------------------------------------------------------
__global__ void __launch_bounds__(256) edge_msg_kernel(
    const float* __restrict__ Xi, const float* __restrict__ Xj,
    const int* __restrict__ idx_i, const int* __restrict__ idx_j,
    const int* __restrict__ agg_idx, const bf16_t* __restrict__ WH,
    const bf16_t* __restrict__ WL, const float* __restrict__ bias,
    float* __restrict__ aggr, int E) {
  __shared__ __align__(32) bf16_t smH[2 * FRAGK];
  __shared__ __align__(32) bf16_t smL[2 * FRAGK];
  int tid = threadIdx.x;
  int w = tid >> 5, lane = tid & 31;
  int h = lane >> 4, mr = lane & 15;
  int base = blockIdx.x * 128 + w * 16;
  int er = base + mr;
  if (er >= E) er = E - 1;
  const float* rowA = Xi + (size_t)idx_i[er] * D128;
  const float* rowB = Xj + (size_t)idx_j[er] * D128;

  v8f acc[8];
#pragma unroll
  for (int i = 0; i < 8; ++i) acc[i] = (v8f)0.0f;

  wmma_block_gemm<8>(rowA, rowB, WH, WL, smH, smL, acc, tid);

#pragma unroll
  for (int nt = 0; nt < 8; ++nt) {
    int col = nt * 16 + mr;
    float bv = bias[col];
#pragma unroll
    for (int r = 0; r < 8; ++r) {
      int e = base + r + 8 * h;  // D layout: M = vgpr + 8*(lane>=16), N = lane&15
      if (e < E) {
        float v = acc[nt][r] + bv;
        v = v > 0.0f ? v : 0.0f;
        atomicAdd(aggr + (size_t)agg_idx[e] * D128 + col, v);
      }
    }
  }
}

// ---------------------------------------------------------------------------
// Dense GEMM: out = [relu](concat[XA,XB] @ W + bias) [+ residual]
// KT=8 -> K=256 (XA first 128, XB second 128); KT=4 -> K=128 (XA only).
// ---------------------------------------------------------------------------
template <int KT, bool RELU>
__global__ void __launch_bounds__(256) dense_gemm_kernel(
    const float* __restrict__ XA, const float* __restrict__ XB,
    const bf16_t* __restrict__ WH, const bf16_t* __restrict__ WL,
    const float* __restrict__ bias, const float* __restrict__ residual,
    float* __restrict__ out, int M) {
  __shared__ __align__(32) bf16_t smH[2 * FRAGK];
  __shared__ __align__(32) bf16_t smL[2 * FRAGK];
  int tid = threadIdx.x;
  int w = tid >> 5, lane = tid & 31;
  int h = lane >> 4, mr = lane & 15;
  int base = blockIdx.x * 128 + w * 16;
  int row = base + mr;
  int rc = row < M ? row : (M - 1);
  const float* rowA = XA + (size_t)rc * D128;
  const float* rowB = (KT == 8) ? (XB + (size_t)rc * D128) : nullptr;

  v8f acc[8];
#pragma unroll
  for (int i = 0; i < 8; ++i) acc[i] = (v8f)0.0f;

  wmma_block_gemm<KT>(rowA, rowB, WH, WL, smH, smL, acc, tid);

#pragma unroll
  for (int nt = 0; nt < 8; ++nt) {
    int col = nt * 16 + mr;
    float bv = bias[col];
#pragma unroll
    for (int r = 0; r < 8; ++r) {
      int m = base + r + 8 * h;
      if (m < M) {
        float v = acc[nt][r] + bv;
        if (RELU) v = v > 0.0f ? v : 0.0f;
        if (residual) v += residual[(size_t)m * D128 + col];
        out[(size_t)m * D128 + col] = v;
      }
    }
  }
}

// gate[f] = dot(F[f,:], Wg) + bg
__global__ void gate_kernel(const float* __restrict__ F, const float* __restrict__ Wg,
                            const float* __restrict__ bg, float* __restrict__ gate,
                            int NF) {
  int f = blockIdx.x * blockDim.x + threadIdx.x;
  if (f >= NF) return;
  const float4* row = reinterpret_cast<const float4*>(F + (size_t)f * D128);
  const float4* wg  = reinterpret_cast<const float4*>(Wg);
  float s = 0.0f;
#pragma unroll
  for (int i = 0; i < 32; ++i) {
    float4 a = row[i], b = wg[i];
    s += a.x * b.x + a.y * b.y + a.z * b.z + a.w * b.w;
  }
  gate[f] = s + bg[0];
}

__device__ __forceinline__ int lower_bound_i32(const int* __restrict__ a, int n, int v) {
  int lo = 0, hi = n;
  while (lo < hi) {
    int mid = (lo + hi) >> 1;
    if (a[mid] < v) lo = mid + 1; else hi = mid;
  }
  return lo;
}

// One block per group: segment softmax over gate + weighted sum of t rows.
__global__ void __launch_bounds__(256) attn_kernel(
    const float* __restrict__ gate, const float* __restrict__ t,
    const int* __restrict__ batch, float* __restrict__ g_att, int NF) {
  __shared__ float red[256];
  __shared__ int srange[2];
  int g = blockIdx.x;
  if (threadIdx.x == 0) {
    srange[0] = lower_bound_i32(batch, NF, g);
    srange[1] = lower_bound_i32(batch, NF, g + 1);
  }
  __syncthreads();
  int s = srange[0], e = srange[1];
  int col = threadIdx.x & 127;
  if (s >= e) {
    if (threadIdx.x < 128) g_att[(size_t)g * D128 + col] = 0.0f;
    return;
  }
  float mx = -__builtin_inff();
  for (int f = s + (int)threadIdx.x; f < e; f += 256) mx = fmaxf(mx, gate[f]);
  red[threadIdx.x] = mx;
  __syncthreads();
  for (int st = 128; st > 0; st >>= 1) {
    if ((int)threadIdx.x < st) red[threadIdx.x] = fmaxf(red[threadIdx.x], red[threadIdx.x + st]);
    __syncthreads();
  }
  mx = red[0];
  __syncthreads();
  float sm = 0.0f;
  for (int f = s + (int)threadIdx.x; f < e; f += 256) sm += expf(gate[f] - mx);
  red[threadIdx.x] = sm;
  __syncthreads();
  for (int st = 128; st > 0; st >>= 1) {
    if ((int)threadIdx.x < st) red[threadIdx.x] += red[threadIdx.x + st];
    __syncthreads();
  }
  float denom = red[0];
  __syncthreads();
  int half = threadIdx.x >> 7;
  float acc = 0.0f;
  for (int f = s + half; f < e; f += 2)
    acc += expf(gate[f] - mx) * t[(size_t)f * D128 + col];
  red[threadIdx.x] = acc;
  __syncthreads();
  if (half == 0)
    g_att[(size_t)g * D128 + col] = (red[col] + red[col + 128]) / denom;
}

// ---------------------------------------------------------------------------
extern "C" void kernel_launch(void* const* d_in, const int* in_sizes, int n_in,
                              void* d_out, int out_size, void* d_ws, size_t ws_size,
                              hipStream_t stream) {
  (void)n_in; (void)out_size; (void)ws_size;
  const float* variables = (const float*)d_in[0];
  const float* factors   = (const float*)d_in[1];
  const int*   edge_index= (const int*)d_in[2];
  const int*   batch_idx = (const int*)d_in[4];
  const float* Wm_vf = (const float*)d_in[5];
  const float* bm_vf = (const float*)d_in[6];
  const float* Wc_vf = (const float*)d_in[7];
  const float* bc_vf = (const float*)d_in[8];
  const float* Wm_fv = (const float*)d_in[9];
  const float* bm_fv = (const float*)d_in[10];
  const float* Wc_fv = (const float*)d_in[11];
  const float* bc_fv = (const float*)d_in[12];
  const float* Wg = (const float*)d_in[13];
  const float* bg = (const float*)d_in[14];
  const float* Wt = (const float*)d_in[15];
  const float* bt = (const float*)d_in[16];
  const float* Wl = (const float*)d_in[17];
  const float* bl = (const float*)d_in[18];

  const int NV = in_sizes[0] / D128;
  const int NF = in_sizes[1] / D128;
  const int E  = in_sizes[2] / 2;
  const int G  = 64, L = 2;
  const int* src_v = edge_index;      // variable index per edge
  const int* dst_f = edge_index + E;  // factor index per edge

  uintptr_t cur = (uintptr_t)d_ws;
  auto alloc = [&](size_t bytes) -> void* {
    uintptr_t p = (cur + 255) & ~(uintptr_t)255;
    cur = p + bytes;
    return (void*)p;
  };
  float* wsV   = (float*)alloc((size_t)NV * D128 * 4);
  float* wsF   = (float*)alloc((size_t)NF * D128 * 4);
  float* aggF  = (float*)alloc((size_t)NF * D128 * 4);  // reused later as t
  float* aggV  = (float*)alloc((size_t)NV * D128 * 4);
  float* gateb = (float*)alloc((size_t)NF * 4);
  float* g_att = (float*)alloc((size_t)G * D128 * 4);

  const size_t FRAG8 = (size_t)8 * FRAGK;  // K=256 packed elements
  const size_t FRAG4 = (size_t)4 * FRAGK;  // K=128 packed elements
  bf16_t *pmvfH[2], *pmvfL[2], *pcvfH[2], *pcvfL[2];
  bf16_t *pmfvH[2], *pmfvL[2], *pcfvH[2], *pcfvL[2];
  for (int l = 0; l < L; ++l) {
    pmvfH[l] = (bf16_t*)alloc(FRAG8 * 2); pmvfL[l] = (bf16_t*)alloc(FRAG8 * 2);
    pcvfH[l] = (bf16_t*)alloc(FRAG8 * 2); pcvfL[l] = (bf16_t*)alloc(FRAG8 * 2);
    pmfvH[l] = (bf16_t*)alloc(FRAG8 * 2); pmfvL[l] = (bf16_t*)alloc(FRAG8 * 2);
    pcfvH[l] = (bf16_t*)alloc(FRAG8 * 2); pcfvL[l] = (bf16_t*)alloc(FRAG8 * 2);
  }
  bf16_t* pWtH = (bf16_t*)alloc(FRAG4 * 2); bf16_t* pWtL = (bf16_t*)alloc(FRAG4 * 2);
  bf16_t* pWlH = (bf16_t*)alloc(FRAG4 * 2); bf16_t* pWlL = (bf16_t*)alloc(FRAG4 * 2);

  const int pk8 = (int)((FRAG8 + 255) / 256), pk4 = (int)((FRAG4 + 255) / 256);
  for (int l = 0; l < L; ++l) {
    pack_w_kernel<<<pk8, 256, 0, stream>>>(Wm_vf + (size_t)l * 256 * D128, pmvfH[l], pmvfL[l], 8);
    pack_w_kernel<<<pk8, 256, 0, stream>>>(Wc_vf + (size_t)l * 256 * D128, pcvfH[l], pcvfL[l], 8);
    pack_w_kernel<<<pk8, 256, 0, stream>>>(Wm_fv + (size_t)l * 256 * D128, pmfvH[l], pmfvL[l], 8);
    pack_w_kernel<<<pk8, 256, 0, stream>>>(Wc_fv + (size_t)l * 256 * D128, pcfvH[l], pcfvL[l], 8);
  }
  pack_w_kernel<<<pk4, 256, 0, stream>>>(Wt, pWtH, pWtL, 4);
  pack_w_kernel<<<pk4, 256, 0, stream>>>(Wl, pWlH, pWlL, 4);  // first 128 rows (g_prev==0)

  hipMemcpyAsync(wsV, variables, (size_t)NV * D128 * 4, hipMemcpyDeviceToDevice, stream);
  hipMemcpyAsync(wsF, factors,   (size_t)NF * D128 * 4, hipMemcpyDeviceToDevice, stream);

  const int eblocks = (E + 127) / 128;
  const int fblocks = (NF + 127) / 128;
  const int vblocks = (NV + 127) / 128;

  for (int l = 0; l < L; ++l) {
    zero_kernel<<<4096, 256, 0, stream>>>(aggF, (long long)NF * D128);
    zero_kernel<<<4096, 256, 0, stream>>>(aggV, (long long)NV * D128);
    // var -> factor messages: concat[factors[dst], variables[src]] -> sum into aggF[dst]
    edge_msg_kernel<<<eblocks, 256, 0, stream>>>(wsF, wsV, dst_f, src_v, dst_f,
                                                 pmvfH[l], pmvfL[l], bm_vf + l * D128,
                                                 aggF, E);
    // factor -> var messages: concat[variables[src], factors[dst]] -> sum into aggV[src]
    edge_msg_kernel<<<eblocks, 256, 0, stream>>>(wsV, wsF, src_v, dst_f, src_v,
                                                 pmfvH[l], pmfvL[l], bm_fv + l * D128,
                                                 aggV, E);
    // node updates (row-exclusive -> safe in place)
    dense_gemm_kernel<8, true><<<fblocks, 256, 0, stream>>>(
        wsF, aggF, pcvfH[l], pcvfL[l], bc_vf + l * D128, nullptr, wsF, NF);
    dense_gemm_kernel<8, true><<<vblocks, 256, 0, stream>>>(
        wsV, aggV, pcfvH[l], pcfvL[l], bc_fv + l * D128, wsV, wsV, NV);
  }

  gate_kernel<<<(NF + 255) / 256, 256, 0, stream>>>(wsF, Wg, bg, gateb, NF);
  float* tbuf = aggF;  // reuse
  dense_gemm_kernel<4, false><<<fblocks, 256, 0, stream>>>(
      wsF, nullptr, pWtH, pWtL, bt, nullptr, tbuf, NF);
  attn_kernel<<<G, 256, 0, stream>>>(gateb, tbuf, batch_idx, g_att, NF);

  float* out_v = (float*)d_out;
  float* out_f = out_v + (size_t)NV * D128;
  float* out_g = out_f + (size_t)NF * D128;
  dense_gemm_kernel<4, true><<<1, 256, 0, stream>>>(
      g_att, nullptr, pWlH, pWlL, bl, nullptr, out_g, G);

  hipMemcpyAsync(out_v, wsV, (size_t)NV * D128 * 4, hipMemcpyDeviceToDevice, stream);
  hipMemcpyAsync(out_f, wsF, (size_t)NF * D128 * 4, hipMemcpyDeviceToDevice, stream);
}
